// NonLocAtt_38001870635960
// MI455X (gfx1250) — matrix operand order repaired and checked by reading, not compiled, not run on hardware
//
#include <hip/hip_runtime.h>
#include <hip/hip_bf16.h>

// MI455X / gfx1250, wave32. Fused non-local attention:
//   prep: H_kv = w_kv*H, h_q = w_q*h  -> f16 WMMA fragments in ws
//   main: per (b, 16-col tile) WG (8 waves): Q fragments register-resident,
//         software-pipelined B/A fragment loads, paired WMMA tiles (fills the
//         WMMA->VALU hazard window), base-2 online softmax, att store (134MB,
//         the HBM-bound part), LDS-staged h_v WMMA chain, w_o + residual.

typedef _Float16 f16;
typedef _Float16 v16h __attribute__((ext_vector_type(16)));
typedef _Float16 v8h  __attribute__((ext_vector_type(8)));
typedef float    v8f  __attribute__((ext_vector_type(8)));

#define B_   4
#define C_   64
#define T_   8
#define N_   1024
#define HID_ 32
#define R_   (T_ * N_)      // 8192 softmax rows
#define LOG2E_ 1.4426950408889634f

// workspace (f16):
//  qA  : [b][ntile=64][lane=32][16]        A-frag layout of h_q (M=n, K=d)
//  kvB : [b][t=8][mtile=64][lane=32][16]   B-frag layout of H_kv (K=d, N=m)
//  kvA : [b][rchunk=256][dtile=2][lane=32][16] A-frag layout of H_kv (M=d, K=row-in-chunk)
#define QA_ELEMS   (B_ * 64 * 32 * 16)
#define KVB_ELEMS  (B_ * T_ * 64 * 32 * 16)
#define KVA_ELEMS  (B_ * 256 * 2 * 32 * 16)

__global__ void prep_kv_kernel(const float* __restrict__ H,
                               const float* __restrict__ w_kv,
                               f16* __restrict__ kvB, f16* __restrict__ kvA) {
  int tid = blockIdx.x * blockDim.x + threadIdx.x;   // 1,048,576 threads
  int n = tid & (N_ - 1);
  int t = (tid >> 10) & (T_ - 1);
  int d = (tid >> 13) & (HID_ - 1);
  int b = tid >> 18;
  const float* hp = H + (((size_t)(b * C_) * T_ + t) * N_ + n);
  const float* wp = w_kv + d * C_;
  float acc = 0.f;
  #pragma unroll 4
  for (int c = 0; c < C_; ++c)
    acc += wp[c] * hp[(size_t)c * (T_ * N_)];
  f16 v = (f16)acc;
  // kvB: B-matrix 32x16 (K=d, N=n%16): lane = (n&15) + 16*((d>>3)&1), elem = 8*(d>>4) + (d&7)
  {
    int mtile = n >> 4;
    int lane  = (n & 15) + (((d >> 3) & 1) << 4);
    int idx   = ((d >> 4) << 3) + (d & 7);
    kvB[((((size_t)(b * T_ + t) * 64 + mtile) * 32 + lane) << 4) + idx] = v;
  }
  // kvA: A-matrix 16x32 (M=d%16, K=row-in-32chunk)
  {
    int r = t * N_ + n;
    int rchunk = r >> 5;
    int kk = r & 31;
    int dtile = d >> 4, Ml = d & 15;
    int lane = Ml + (((kk >> 3) & 1) << 4);
    int idx  = ((kk >> 4) << 3) + (kk & 7);
    kvA[((((size_t)(b * 256 + rchunk) * 2 + dtile) * 32 + lane) << 4) + idx] = v;
  }
}

__global__ void prep_q_kernel(const float* __restrict__ h,
                              const float* __restrict__ w_q,
                              f16* __restrict__ qA) {
  int tid = blockIdx.x * blockDim.x + threadIdx.x;   // 131,072 threads
  int n = tid & (N_ - 1);
  int d = (tid >> 10) & (HID_ - 1);
  int b = tid >> 15;
  const float* hp = h + (size_t)(b * C_) * N_ + n;
  const float* wp = w_q + d * C_;
  float acc = 0.f;
  #pragma unroll 4
  for (int c = 0; c < C_; ++c)
    acc += wp[c] * hp[(size_t)c * N_];
  f16 v = (f16)acc;
  // qA: A-matrix 16x32 (M=n%16, K=d)
  int ntile = n >> 4, Ml = n & 15;
  int lane = Ml + (((d >> 3) & 1) << 4);
  int idx  = ((d >> 4) << 3) + (d & 7);
  qA[(((size_t)(b * 64 + ntile) * 32 + lane) << 4) + idx] = v;
}

__global__ __launch_bounds__(256)
void nonloc_att_kernel(const f16* __restrict__ qA, const f16* __restrict__ kvB,
                       const f16* __restrict__ kvA,
                       const float* __restrict__ h_in, const float* __restrict__ w_o,
                       float* __restrict__ out1, float* __restrict__ att) {
  __shared__ f16   att_s[8][16][48];   // per-wave staging, 96B rows -> 32B aligned
  __shared__ float statmx[8][32];
  __shared__ float statsm[8][32];
  __shared__ float colmax_s[16];
  __shared__ float colinv_s[16];
  __shared__ float hvred[8][2][32][8];
  __shared__ float hvfin[32][16];

  const int b     = blockIdx.x >> 6;
  const int mtile = blockIdx.x & 63;
  const int m0    = mtile << 4;
  const int tid   = threadIdx.x;
  const int wid   = tid >> 5;          // 8 waves
  const int lane  = tid & 31;
  const int ncol  = lane & 15;         // column within 16-wide tile (C/B frag N)
  const int half  = lane >> 4;

  auto ldB = [&](int t) -> v16h {      // score B-fragment for timestep t
    return *(const v16h*)(kvB + ((((size_t)(b * T_ + t) * 64 + mtile) * 32 + lane) << 4));
  };
  auto ldA = [&](int rchunk, int dt) -> v16h {  // h_v A-fragment
    return *(const v16h*)(kvA + ((((size_t)(b * 256 + rchunk) * 2 + dt) * 32 + lane) << 4));
  };

  // Q fragments for this wave's 8 row tiles: register-resident for the whole
  // kernel (64 VGPRs), reused across 8 t-steps x 2 passes.
  v16h aQf[8];
  #pragma unroll
  for (int rti = 0; rti < 8; ++rti)
    aQf[rti] = *(const v16h*)(qA + (((size_t)(b * 64 + wid * 8 + rti) * 32 + lane) << 4));

  // ---------------- pass 1: per-column online max / sum-exp (base-2 domain) --
  float mx = -3.0e38f, sm = 0.f;
  v16h bK = ldB(0);
  for (int t = 0; t < T_; ++t) {
    v16h bKn = bK;
    if (t < T_ - 1) bKn = ldB(t + 1);          // pipeline next timestep's B-frag
    #pragma unroll
    for (int rtp = 0; rtp < 4; ++rtp) {        // paired tiles: independent accums
      int rtA = wid * 8 + 2 * rtp;
      int rtB = rtA + 1;
      v8f cfA = {}, cfB = {};
      cfA = __builtin_amdgcn_wmma_f32_16x16x32_f16(false, aQf[2 * rtp],     false, bK,
                                                   (short)0, cfA, false, false);
      cfB = __builtin_amdgcn_wmma_f32_16x16x32_f16(false, aQf[2 * rtp + 1], false, bK,
                                                   (short)0, cfB, false, false);
      float va[8], vb[8];
      #pragma unroll
      for (int j = 0; j < 8; ++j) { va[j] = cfA[j] * LOG2E_; vb[j] = cfB[j] * LOG2E_; }
      if (rtA == mtile) {                      // wave-uniform diagonal guards
        #pragma unroll
        for (int j = 0; j < 8; ++j) if ((j + 8 * half) == ncol) va[j] = 0.f;
      }
      if (rtB == mtile) {
        #pragma unroll
        for (int j = 0; j < 8; ++j) if ((j + 8 * half) == ncol) vb[j] = 0.f;
      }
      // one online update per 16 elements
      float tmax = fmaxf(
          fmaxf(fmaxf(fmaxf(va[0], va[1]), fmaxf(va[2], va[3])),
                fmaxf(fmaxf(va[4], va[5]), fmaxf(va[6], va[7]))),
          fmaxf(fmaxf(fmaxf(vb[0], vb[1]), fmaxf(vb[2], vb[3])),
                fmaxf(fmaxf(vb[4], vb[5]), fmaxf(vb[6], vb[7]))));
      float nm = fmaxf(mx, tmax);
      float tsum = 0.f;
      #pragma unroll
      for (int j = 0; j < 8; ++j)
        tsum += __builtin_amdgcn_exp2f(va[j] - nm) + __builtin_amdgcn_exp2f(vb[j] - nm);
      sm = sm * __builtin_amdgcn_exp2f(mx - nm) + tsum;
      mx = nm;
    }
    bK = bKn;
  }
  statmx[wid][lane] = mx;
  statsm[wid][lane] = sm;
  __syncthreads();
  if (tid < 16) {
    float M = -3.0e38f;
    #pragma unroll
    for (int w = 0; w < 8; ++w)
      M = fmaxf(M, fmaxf(statmx[w][tid], statmx[w][tid + 16]));
    float S = 0.f;
    #pragma unroll
    for (int w = 0; w < 8; ++w) {
      S += statsm[w][tid]      * __builtin_amdgcn_exp2f(statmx[w][tid]      - M);
      S += statsm[w][tid + 16] * __builtin_amdgcn_exp2f(statmx[w][tid + 16] - M);
    }
    colmax_s[tid] = M;
    colinv_s[tid] = 1.f / S;
  }
  __syncthreads();
  const float cmax = colmax_s[ncol];   // base-2 domain
  const float cinv = colinv_s[ncol];

  // ---------------- pass 2: recompute scores -> att, accumulate h_v -----------
  v8f hv0 = {}, hv1 = {};
  bK = ldB(0);
  v16h a0 = ldA(wid * 4, 0);
  v16h a1 = ldA(wid * 4, 1);
  for (int it = 0; it < 32; ++it) {            // flattened (t, rp) loop
    const int t  = it >> 2;
    const int rp = it & 3;
    const int rt0 = wid * 8 + rp * 2;
    // pipeline next iteration's fragments
    v16h bKn = bK, a0n = a0, a1n = a1;
    if (it < 31) {
      int tn = (it + 1) >> 2, rpn = (it + 1) & 3;
      if (tn != t) bKn = ldB(tn);
      int rcn = tn * 32 + wid * 4 + rpn;
      a0n = ldA(rcn, 0);
      a1n = ldA(rcn, 1);
    }
    // paired score tiles
    v8f cfA = {}, cfB = {};
    cfA = __builtin_amdgcn_wmma_f32_16x16x32_f16(false, aQf[rp * 2],     false, bK,
                                                 (short)0, cfA, false, false);
    cfB = __builtin_amdgcn_wmma_f32_16x16x32_f16(false, aQf[rp * 2 + 1], false, bK,
                                                 (short)0, cfB, false, false);
    float va[8], vb[8];
    #pragma unroll
    for (int j = 0; j < 8; ++j) { va[j] = cfA[j]; vb[j] = cfB[j]; }
    if (rt0 == mtile) {
      #pragma unroll
      for (int j = 0; j < 8; ++j) if ((j + 8 * half) == ncol) va[j] = 0.f;
    }
    if (rt0 + 1 == mtile) {
      #pragma unroll
      for (int j = 0; j < 8; ++j) if ((j + 8 * half) == ncol) vb[j] = 0.f;
    }
    float ea[8], eb[8];
    #pragma unroll
    for (int j = 0; j < 8; ++j) {              // fma -> exp2 -> scale
      ea[j] = __builtin_amdgcn_exp2f(fmaf(va[j], LOG2E_, -cmax)) * cinv;
      eb[j] = __builtin_amdgcn_exp2f(fmaf(vb[j], LOG2E_, -cmax)) * cinv;
    }
    // att stores: rows rt*16 + 8*half + j, column m0+ncol
    float* apA = att + ((size_t)b * R_ + (size_t)t * N_ + rt0 * 16 + 8 * half) * N_
                     + (m0 + ncol);
    float* apB = apA + (size_t)16 * N_;
    #pragma unroll
    for (int j = 0; j < 8; ++j) {
      apA[(size_t)j * N_] = ea[j];
      apB[(size_t)j * N_] = eb[j];
    }
    // staging for the h_v B-fragment (2x ds_store_b128)
    v8h packA, packB;
    #pragma unroll
    for (int j = 0; j < 8; ++j) { packA[j] = (f16)ea[j]; packB[j] = (f16)eb[j]; }
    *(v8h*)&att_s[wid][ncol][16 * half]      = packA;
    *(v8h*)&att_s[wid][ncol][16 * half + 8] = packB;
    // h_v += KV[:, chunk] x att[chunk, :] ; chunk = 32 rows (rt0, rt0+1)
    v16h bAtt = *(const v16h*)&att_s[wid][ncol][half * 16];  // 2x ds_read_b128
    hv0 = __builtin_amdgcn_wmma_f32_16x16x32_f16(false, a0, false, bAtt,
                                                 (short)0, hv0, false, false);
    hv1 = __builtin_amdgcn_wmma_f32_16x16x32_f16(false, a1, false, bAtt,
                                                 (short)0, hv1, false, false);
    bK = bKn; a0 = a0n; a1 = a1n;
  }

  // reduce h_v across the 8 waves
  #pragma unroll
  for (int j = 0; j < 8; ++j) {
    hvred[wid][0][lane][j] = hv0[j];
    hvred[wid][1][lane][j] = hv1[j];
  }
  __syncthreads();
  for (int idx = tid; idx < HID_ * 16; idx += 256) {
    int d = idx >> 4, m = idx & 15;
    int dt = d >> 4, Ml = d & 15;
    int jj = Ml & 7, hh = Ml >> 3;
    int ln = m + 16 * hh;
    float s = 0.f;
    #pragma unroll
    for (int w = 0; w < 8; ++w) s += hvred[w][dt][ln][jj];
    hvfin[d][m] = s;
  }
  __syncthreads();

  // out1 = h + w_o * h_v  (residual + channel mix)
  for (int idx = tid; idx < C_ * 16; idx += 256) {
    int c = idx >> 4, m = idx & 15;
    float s = h_in[(size_t)(b * C_ + c) * N_ + m0 + m];
    #pragma unroll 4
    for (int d = 0; d < HID_; ++d) s += w_o[c * HID_ + d] * hvfin[d][m];
    out1[(size_t)(b * C_ + c) * N_ + m0 + m] = s;
  }
}

extern "C" void kernel_launch(void* const* d_in, const int* in_sizes, int n_in,
                              void* d_out, int out_size, void* d_ws, size_t ws_size,
                              hipStream_t stream) {
  (void)in_sizes; (void)n_in; (void)out_size; (void)ws_size;
  const float* H    = (const float*)d_in[0];
  const float* h    = (const float*)d_in[1];
  const float* w_q  = (const float*)d_in[2];
  const float* w_kv = (const float*)d_in[3];
  const float* w_o  = (const float*)d_in[4];

  float* out1 = (float*)d_out;                 // [B,C,Hh,Ww] = 262144 floats
  float* attO = out1 + (size_t)B_ * C_ * N_;   // [B,T*N,N]   = 33,554,432 floats

  f16* qA  = (f16*)d_ws;                       // 256 KB
  f16* kvB = qA + QA_ELEMS;                    // 2 MB
  f16* kvA = kvB + KVB_ELEMS;                  // 2 MB

  prep_kv_kernel<<<(B_ * HID_ * T_ * N_) / 256, 256, 0, stream>>>(H, w_kv, kvB, kvA);
  prep_q_kernel<<<(B_ * HID_ * N_) / 256, 256, 0, stream>>>(h, w_q, qA);
  nonloc_att_kernel<<<B_ * 64, 256, 0, stream>>>(qA, kvB, kvA, h, w_o, out1, attO);
}
